// Guassian_Attention_40183714021647
// MI455X (gfx1250) — matrix operand order repaired and checked by reading, not compiled, not run on hardware
//
#include <hip/hip_runtime.h>
#include <hip/hip_bf16.h>
#include <math.h>

typedef __attribute__((ext_vector_type(16))) __bf16 v16bf;
typedef __attribute__((ext_vector_type(8)))  float  v8f;

// B=8, H=8, L=1024, d=64
#define BH_   64
#define L_    1024
#define D_    64
#define LL_   1048576     // L*L
#define LD_   65536       // L*d
#define ADJR_ 1023        // L-1
#define ADJS_ 1046529     // (L-1)*(L-1)

#define USE_ASYNC 1

// ---------------------------------------------------------------------------
// Kernel 1: Pi[b,h,q] = sigmoid( tanh(q @ W1 + b1) @ W2 + b2 ) * L
// ---------------------------------------------------------------------------
__global__ void __launch_bounds__(256) pi_kernel(
    const float* __restrict__ q,
    const float* __restrict__ W1, const float* __restrict__ b1,
    const float* __restrict__ W2, const float* __restrict__ b2,
    float* __restrict__ Pi)
{
    __shared__ float w1s[64 * 64];
    __shared__ float w2s[64];
    __shared__ float b1s[64];
    for (int i = threadIdx.x; i < 4096; i += 256) w1s[i] = W1[i];
    if (threadIdx.x < 64) {
        w2s[threadIdx.x] = W2[threadIdx.x];
        b1s[threadIdx.x] = b1[threadIdx.x];
    }
    __syncthreads();

    const int row = blockIdx.x * 256 + threadIdx.x;  // 0 .. 65535
    const float* qr = q + (size_t)row * 64;
    float qv[64];
#pragma unroll
    for (int i = 0; i < 64; ++i) qv[i] = qr[i];

    float acc2 = b2[0];
    for (int j = 0; j < 64; ++j) {
        float s = b1s[j];
#pragma unroll
        for (int i = 0; i < 64; ++i) s += qv[i] * w1s[i * 64 + j];
        acc2 += tanhf(s) * w2s[j];
    }
    Pi[row] = 1024.0f / (1.0f + __expf(-acc2));   // sigmoid * L
}

// ---------------------------------------------------------------------------
// Async global->LDS copy of one 64x64 fp32 tile (16 KB), 256 threads.
// Each thread copies 64 contiguous bytes with 4 GLOBAL_LOAD_ASYNC_TO_LDS_B128.
// Tracked by ASYNCcnt (4 outstanding ops per thread per tile).
// ---------------------------------------------------------------------------
__device__ __forceinline__ void async_tile_issue(const float* __restrict__ gsrc,
                                                 float* stage, int tid)
{
    const float* g = gsrc + tid * 16;
#if USE_ASYNC
    // low 32 bits of a generic shared pointer = LDS byte address
    unsigned l = (unsigned)(size_t)(const void*)(stage + tid * 16);
    asm volatile(
        "global_load_async_to_lds_b128 %0, %1, off\n\t"
        "global_load_async_to_lds_b128 %0, %1, off offset:16\n\t"
        "global_load_async_to_lds_b128 %0, %1, off offset:32\n\t"
        "global_load_async_to_lds_b128 %0, %1, off offset:48"
        :: "v"(l), "v"(g) : "memory");
#else
    float* d = stage + tid * 16;
#pragma unroll
    for (int j = 0; j < 16; ++j) d[j] = g[j];
#endif
}

__device__ __forceinline__ void async_wait(int pending)
{
#if USE_ASYNC
    if (pending)
        asm volatile("s_wait_asynccnt 4" ::: "memory");
    else
        asm volatile("s_wait_asynccnt 0" ::: "memory");
#endif
}

// ---------------------------------------------------------------------------
// Kernel 2: fused attention for one (b,h) and a 64-row q tile.
// ---------------------------------------------------------------------------
__global__ void __launch_bounds__(256) attn_kernel(
    const float* __restrict__ query, const float* __restrict__ key,
    const float* __restrict__ value, const int* __restrict__ maskp,
    const int* __restrict__ adjp,   const float* __restrict__ Pi,
    float* __restrict__ outp,       float* __restrict__ pattn)
{
    extern __shared__ __align__(16) char smem[];
    __bf16* sc   = (__bf16*)smem;                   // 64x1024 scores/probs bf16 (128 KB)
    __bf16* qb   = sc + 64 * 1024;                  // 64x64 Q bf16 (8 KB)
    __bf16* kvb  = qb + 64 * 64;                    // 64x64 K / V^T bf16 (8 KB)
    float*  stg  = (float*)(kvb + 64 * 64);         // 2 x 64x64 fp32 staging (32 KB)
    unsigned char* mk = (unsigned char*)(stg + 2 * 4096); // 64x1024 fused mask (64 KB)
    float*  pis  = (float*)(mk + 64 * 1024);        // 64 Pi values

    const int qt   = blockIdx.x;          // 0..15 q tile
    const int bh   = blockIdx.y;          // 0..63
    const int b    = bh >> 3;             // batch (H = 8)
    const int q0   = qt * 64;
    const int tid  = threadIdx.x;
    const int lane = tid & 31;
    const int wave = tid >> 5;
    const int lm   = lane & 15;           // position within half-wave
    const int g    = lane >> 4;           // half-wave group (K-split select)

    const float* Kbase = key   + (size_t)bh * LD_;
    const float* Vbase = value + (size_t)bh * LD_;

    // kick async load of K tile 0 first: overlaps mask fusion below
    async_tile_issue(Kbase, stg, tid);

    // -------- load Q tile (fp32 -> bf16) and Pi segment --------
    const float* Qg = query + (size_t)bh * LD_ + (size_t)q0 * 64;
    for (int i = tid; i < 4096; i += 256) qb[i] = (__bf16)Qg[i];
    if (tid < 64) pis[tid] = Pi[bh * L_ + q0 + tid];

    // -------- fuse mask * adj_p for this q band into LDS bytes --------
    for (int i = tid; i < 65536; i += 256) {
        const int row = i >> 10, col = i & 1023;
        const int qg  = q0 + row;
        const int mv  = maskp[(size_t)b * LL_ + (size_t)qg * L_ + col];
        const int av  = (qg < ADJR_ && col < ADJR_)
                          ? adjp[(size_t)b * ADJS_ + (size_t)qg * ADJR_ + col]
                          : 1;
        mk[i] = (unsigned char)((mv != 0) & (av != 0));
    }
    __syncthreads();

    // ================= Phase 1: scores =================
    for (int kt = 0; kt < 16; ++kt) {
        const int buf = kt & 1;
        if (kt + 1 < 16) {
            async_tile_issue(Kbase + (size_t)(kt + 1) * 4096,
                             stg + (buf ^ 1) * 4096, tid);
            async_wait(1);
        } else {
            async_wait(0);
        }
        __syncthreads();               // staging tile kt visible to all waves

        // convert staged fp32 tile -> bf16 operand tile
        {
            const float* s = stg + buf * 4096;
            const int base = tid * 16;
#pragma unroll
            for (int j = 0; j < 16; ++j) kvb[base + j] = (__bf16)s[base + j];
        }
        __syncthreads();

        for (int t = wave; t < 16; t += 8) {
            const int mi = t >> 2, ni = t & 3;
            v8f acc = {};
            const int arow = mi * 16 + lm;    // Q row within tile
            const int brow = ni * 16 + lm;    // K row (= column of K^T)
#pragma unroll
            for (int kc = 0; kc < 2; ++kc) {
                union { v16bf v; __bf16 e[16]; } a, bb;
                const __bf16* ab = qb + arow * 64 + kc * 32;
#pragma unroll
                for (int i = 0; i < 8; ++i) {            // A: contiguous K, k = 16g + s
                    a.e[i]     = ab[g * 16 + i];
                    a.e[8 + i] = ab[g * 16 + 8 + i];
                }
                const __bf16* bp = kvb + brow * 64 + kc * 32 + g * 16;
#pragma unroll
                for (int i = 0; i < 16; ++i) bb.e[i] = bp[i];
                acc = __builtin_amdgcn_wmma_f32_16x16x32_bf16(
                    false, a.v, false, bb.v, (short)0, acc, false, false);
            }
            // epilogue: scale, Gaussian bias, fused mask -> LDS bf16
            const int col_l = ni * 16 + lm;
            const int col   = kt * 64 + col_l;
#pragma unroll
            for (int r = 0; r < 8; ++r) {
                const int row_l = mi * 16 + g * 8 + r;   // D layout: M = r + g*8
                float s  = acc[r] * 0.125f;              // 1/sqrt(64)
                float dj = (float)col - pis[row_l];
                s -= dj * dj * (1.0f / 9.0f);            // (D/2)^2 = 9
                if (!mk[row_l * 1024 + col]) s = -1.0e9f;
                sc[row_l * 1024 + col] = (__bf16)s;
            }
        }
        __syncthreads();
    }

    // prefetch V tile 0 asynchronously; it streams in during softmax
    async_tile_issue(Vbase, stg, tid);

    // ================= Phase 2: softmax + p_attn writeout =================
    for (int i = 0; i < 8; ++i) {
        const int row = wave * 8 + i;
        const __bf16* srow = sc + row * 1024;
        float m = -3.0e38f;
        for (int c = lane; c < 1024; c += 32)
            m = fmaxf(m, (float)srow[c]);
#pragma unroll
        for (int o = 16; o > 0; o >>= 1) m = fmaxf(m, __shfl_xor(m, o, 32));
        float sum = 0.0f;
        for (int c = lane; c < 1024; c += 32)
            sum += __expf((float)srow[c] - m);
#pragma unroll
        for (int o = 16; o > 0; o >>= 1) sum += __shfl_xor(sum, o, 32);
        const float inv = 1.0f / sum;

        float*  prow = pattn + (size_t)bh * LL_ + (size_t)(q0 + row) * L_;
        __bf16* wrow = sc + row * 1024;
        for (int c = lane; c < 1024; c += 32) {
            float p = __expf((float)srow[c] - m) * inv;
            prow[c] = p;                 // fp32 p_attn (second output)
            wrow[c] = (__bf16)p;         // bf16 probs for P*V
        }
    }

    // ================= Phase 3: out = P * V =================
    v8f oacc[2] = {{}, {}};
    for (int kt = 0; kt < 16; ++kt) {
        const int buf = kt & 1;
        if (kt + 1 < 16) {
            async_tile_issue(Vbase + (size_t)(kt + 1) * 4096,
                             stg + (buf ^ 1) * 4096, tid);
            async_wait(1);
        } else {
            async_wait(0);
        }
        __syncthreads();               // also orders softmax sc writes vs reads (kt==0)

        // convert staged fp32 V tile -> bf16, transposed: kvb[d][k]
        {
            const float* s = stg + buf * 4096;
            const int base = tid * 16;
#pragma unroll
            for (int j = 0; j < 16; ++j) {
                const int i = base + j;
                const int r = i >> 6, c = i & 63;
                kvb[c * 64 + r] = (__bf16)s[i];
            }
        }
        __syncthreads();

        int ti = 0;
        for (int t = wave; t < 16; t += 8, ++ti) {
            const int mi = t >> 2, ni = t & 3;
            const int arow = mi * 16 + lm;
            const int bn   = ni * 16 + lm;    // output column of d
#pragma unroll
            for (int kc = 0; kc < 2; ++kc) {
                union { v16bf v; __bf16 e[16]; } a, bb;
                const __bf16* ab = sc + arow * 1024 + kt * 64 + kc * 32;
#pragma unroll
                for (int i2 = 0; i2 < 8; ++i2) {
                    a.e[i2]     = ab[g * 16 + i2];
                    a.e[8 + i2] = ab[g * 16 + 8 + i2];
                }
                const __bf16* bp = kvb + bn * 64 + kc * 32 + g * 16;
#pragma unroll
                for (int i2 = 0; i2 < 16; ++i2) bb.e[i2] = bp[i2];
                oacc[ti] = __builtin_amdgcn_wmma_f32_16x16x32_bf16(
                    false, a.v, false, bb.v, (short)0, oacc[ti], false, false);
            }
        }
        __syncthreads();
    }

    // writeback out tile
    {
        int ti = 0;
        for (int t = wave; t < 16; t += 8, ++ti) {
            const int mi = t >> 2, ni = t & 3;
            const int col = ni * 16 + lm;
#pragma unroll
            for (int r = 0; r < 8; ++r) {
                const int row_l = mi * 16 + g * 8 + r;
                outp[(size_t)bh * LD_ + (size_t)(q0 + row_l) * 64 + col] = oacc[ti][r];
            }
        }
    }
}

// ---------------------------------------------------------------------------
extern "C" void kernel_launch(void* const* d_in, const int* in_sizes, int n_in,
                              void* d_out, int out_size, void* d_ws, size_t ws_size,
                              hipStream_t stream)
{
    const float* query = (const float*)d_in[0];
    const float* key   = (const float*)d_in[1];
    const float* value = (const float*)d_in[2];
    const int*   mask  = (const int*)d_in[3];
    const int*   adj   = (const int*)d_in[4];
    const float* W1    = (const float*)d_in[5];
    const float* b1    = (const float*)d_in[6];
    const float* W2    = (const float*)d_in[7];
    const float* b2    = (const float*)d_in[8];

    float* out   = (float*)d_out;
    float* pattn = out + (size_t)BH_ * L_ * D_;   // tuple: (out, p_attn)
    float* Pi    = (float*)d_ws;                  // 65536 floats = 256 KB

    pi_kernel<<<256, 256, 0, stream>>>(query, W1, b1, W2, b2, Pi);

    const size_t smem = (size_t)(64 * 1024 + 64 * 64 + 64 * 64) * sizeof(__bf16)
                      + 2 * 4096 * sizeof(float)   // async staging
                      + 64 * 1024                  // fused mask bytes
                      + 64 * sizeof(float);        // ~246 KB of 320 KB/WGP LDS
    attn_kernel<<<dim3(16, 64), 256, smem, stream>>>(
        query, key, value, mask, adj, Pi, out, pattn);
}
